// Emb_Attention_10118942949625
// MI455X (gfx1250) — matrix operand order repaired and checked by reading, not compiled, not run on hardware
//
#include <hip/hip_runtime.h>

// ---------------------------------------------------------------------------
// Types and helpers
// ---------------------------------------------------------------------------
typedef __attribute__((ext_vector_type(16))) __bf16          v16bf;
typedef __attribute__((ext_vector_type(8)))  float           v8f;
typedef __attribute__((ext_vector_type(8)))  unsigned short  ushort8;

struct U4x2 { uint4 a, b; };

static __device__ __forceinline__ v16bf make_frag(uint4 lo, uint4 hi) {
  U4x2 p; p.a = lo; p.b = hi;
  return __builtin_bit_cast(v16bf, p);
}

// f32 -> bf16 round-to-nearest-even (bit trick)
static __device__ __forceinline__ unsigned short f2bf(float f) {
  unsigned u = __builtin_bit_cast(unsigned, f);
  u += 0x7FFFu + ((u >> 16) & 1u);
  return (unsigned short)(u >> 16);
}

static __device__ __forceinline__ v8f wmma_bf16(v16bf a, v16bf b, v8f c) {
  // 8 args: (neg_a, A, neg_b, B, c_mod, C, reuse_a, reuse_b)
  return __builtin_amdgcn_wmma_f32_16x16x32_bf16(false, a, false, b,
                                                 (short)0, c, false, false);
}

// Async DMA: global -> LDS, 16 bytes per lane, tracked by ASYNCcnt.
// lds_off = LDS byte offset (low 32 bits of a generic __shared__ address;
// the LDS aperture only occupies addr[63:32], so the truncation is exact).
static __device__ __forceinline__ void async_load_b128(unsigned lds_off,
                                                       const void* gptr) {
  asm volatile("global_load_async_to_lds_b128 %0, %1, off"
               :: "v"(lds_off), "v"(gptr)
               : "memory");
}

static __device__ __forceinline__ void wait_async0() {
#if __has_builtin(__builtin_amdgcn_s_wait_asynccnt)
  __builtin_amdgcn_s_wait_asynccnt(0);
#else
  asm volatile("s_wait_asynccnt 0x0" ::: "memory");
#endif
}

// A fragment (16x32 bf16): lane<16 -> K 0..7 & 16..23 ; lane>=16 -> K 8..15 & 24..31
static __device__ __forceinline__ v16bf frag_a_lds(const unsigned short* base,
                                                   int stride, int m0, int k0, int lane) {
  const unsigned short* p = base + (m0 + (lane & 15)) * stride + k0 + ((lane & 16) ? 8 : 0);
  uint4 lo = *(const uint4*)p;
  uint4 hi = *(const uint4*)(p + 16);
  return make_frag(lo, hi);
}

// B fragment (32x16 bf16): lane<16 -> K 0..7 & 16..23 of col n ; lane>=16 -> K 8..15 & 24..31
static __device__ __forceinline__ v16bf frag_b_lds(const unsigned short* base,
                                                   int stride, int n0, int k0, int lane) {
  const unsigned short* p = base + (n0 + (lane & 15)) * stride + k0 + ((lane & 16) ? 8 : 0);
  uint4 lo = *(const uint4*)p;
  uint4 hi = *(const uint4*)(p + 16);
  return make_frag(lo, hi);
}

// ---------------------------------------------------------------------------
// f32 -> bf16 conversion kernel (sizes are multiples of 2048)
// ---------------------------------------------------------------------------
__global__ __launch_bounds__(256) void cvt_f32_bf16_kernel(
    const float* __restrict__ in, unsigned short* __restrict__ out, int n) {
  int i = (blockIdx.x * 256 + threadIdx.x) * 8;
  if (i >= n) return;
  float4 a = *(const float4*)(in + i);
  float4 b = *(const float4*)(in + i + 4);
  ushort8 o;
  o[0] = f2bf(a.x); o[1] = f2bf(a.y); o[2] = f2bf(a.z); o[3] = f2bf(a.w);
  o[4] = f2bf(b.x); o[5] = f2bf(b.y); o[6] = f2bf(b.z); o[7] = f2bf(b.w);
  *(ushort8*)(out + i) = o;
}

// ---------------------------------------------------------------------------
// Tiled bf16 WMMA GEMM:  C[m,n] = sum_k A[m,k] * W[n,k]   (C = A @ W^T)
// Block tile 128x128, K-step 32, 8 waves each computing 4x2 16x16 tiles.
// Staging uses async global->LDS DMA (ASYNCcnt), double buffered.
// If Cf != nullptr: f32 output with bias; else bf16 output.
// ---------------------------------------------------------------------------
#define LDSTRIDE 40  // padded bf16 row stride -> conflict-free ds_load_b128

__global__ __launch_bounds__(256) void gemm_bf16_kernel(
    const unsigned short* __restrict__ A, int lda,
    const unsigned short* __restrict__ W, int K,
    unsigned short* __restrict__ Cb, float* __restrict__ Cf,
    int ldc, const float* __restrict__ bias) {
  __shared__ unsigned short As[2][128 * LDSTRIDE];
  __shared__ unsigned short Bs[2][128 * LDSTRIDE];

  const int tid  = threadIdx.x;
  const int lane = tid & 31;
  const int wid  = tid >> 5;
  const int rw   = wid >> 2;   // 0..1  (row wave)
  const int cw   = wid & 3;    // 0..3  (col wave)
  const long mBlock = (long)blockIdx.y * 128;
  const long nBlock = (long)blockIdx.x * 128;

  v8f zf = {};
  v8f acc[4][2];
#pragma unroll
  for (int i = 0; i < 4; ++i)
#pragma unroll
    for (int j = 0; j < 2; ++j) acc[i][j] = zf;

  // 128 rows x 32 k (bf16) per matrix = 512 x 16B chunks; 2 per thread each.
  auto stage_async = [&](int bufI, int k0) {
#pragma unroll
    for (int s = 0; s < 2; ++s) {
      int c    = tid + s * 256;
      int row  = c >> 2;
      int koff = (c & 3) * 8;
      async_load_b128((unsigned)(size_t)&As[bufI][row * LDSTRIDE + koff],
                      A + (mBlock + row) * (long)lda + k0 + koff);
      async_load_b128((unsigned)(size_t)&Bs[bufI][row * LDSTRIDE + koff],
                      W + (nBlock + row) * (long)K + k0 + koff);
    }
  };

  const int nK = K / 32;
  stage_async(0, 0);
  int buf = 0;
  for (int ks = 0; ks < nK; ++ks) {
    wait_async0();       // our async DMAs into the buffer being published
    __syncthreads();     // everyone's DMAs done -> buffer readable
    if (ks + 1 < nK) stage_async(buf ^ 1, (ks + 1) * 32);
    v16bf af[4], bfr[2];
#pragma unroll
    for (int i = 0; i < 4; ++i)
      af[i] = frag_a_lds(&As[buf][0], LDSTRIDE, rw * 64 + i * 16, 0, lane);
#pragma unroll
    for (int j = 0; j < 2; ++j)
      bfr[j] = frag_b_lds(&Bs[buf][0], LDSTRIDE, cw * 32 + j * 16, 0, lane);
#pragma unroll
    for (int i = 0; i < 4; ++i)
#pragma unroll
      for (int j = 0; j < 2; ++j)
        acc[i][j] = wmma_bf16(af[i], bfr[j], acc[i][j]);
    buf ^= 1;
  }

  // C layout: v8f elem e -> row = e + (lane&16 ? 8:0), col = lane&15
  const int mh = (lane & 16) ? 8 : 0;
  const int nc = lane & 15;
  if (Cf) {
#pragma unroll
    for (int i = 0; i < 4; ++i)
#pragma unroll
      for (int j = 0; j < 2; ++j) {
        long col = nBlock + cw * 32 + j * 16 + nc;
        float bv = bias[col];
#pragma unroll
        for (int e = 0; e < 8; ++e) {
          long row = mBlock + rw * 64 + i * 16 + e + mh;
          Cf[row * (long)ldc + col] = acc[i][j][e] + bv;
        }
      }
  } else {
#pragma unroll
    for (int i = 0; i < 4; ++i)
#pragma unroll
      for (int j = 0; j < 2; ++j) {
        long col = nBlock + cw * 32 + j * 16 + nc;
#pragma unroll
        for (int e = 0; e < 8; ++e) {
          long row = mBlock + rw * 64 + i * 16 + e + mh;
          Cb[row * (long)ldc + col] = f2bf(acc[i][j][e]);
        }
      }
  }
}

// ---------------------------------------------------------------------------
// Channel attention, one block per (b,h).
// qkv layout: [B*N, 1536] bf16 ; cols 0:512 = q, 512:1024 = k, 1024:1536 = v,
// each as (h*64 + d). Output overwrites the q columns of this head.
// ---------------------------------------------------------------------------
__global__ __launch_bounds__(256) void chan_attn_kernel(unsigned short* __restrict__ qkv) {
  __shared__ unsigned short Sq[2][64 * LDSTRIDE];  // q^T chunk [d][n]
  __shared__ unsigned short Sk[2][64 * LDSTRIDE];  // k^T chunk [e][n]
  __shared__ float          Sf[64][64];            // scores f32
  __shared__ unsigned short Sab[64 * 72];          // softmaxed attn bf16 [d][e]

  const int tid  = threadIdx.x;
  const int lane = tid & 31;
  const int wid  = tid >> 5;
  const int b    = blockIdx.x >> 3;
  const int h    = blockIdx.x & 7;

  unsigned short*       qb  = qkv + (size_t)b * 4096 * 1536 + h * 64;
  const unsigned short* kb  = qb + 512;
  const unsigned short* vbp = qb + 1024;

  v8f zf = {};
  v8f acc[2]; acc[0] = zf; acc[1] = zf;

  // Stage 32 rows of q/k for this head, transposed into LDS ([d][n]).
  auto stage = [&](int bufI, int n0) {
    int n  = tid >> 3;   // 0..31
    int dg = tid & 7;    // 0..7 (group of 8 d-channels)
    ushort8 vq = *(const ushort8*)(qb + (size_t)(n0 + n) * 1536 + dg * 8);
    ushort8 vk = *(const ushort8*)(kb + (size_t)(n0 + n) * 1536 + dg * 8);
#pragma unroll
    for (int z = 0; z < 8; ++z)
      Sq[bufI][(dg * 8 + z) * LDSTRIDE + n] = vq[z];
#pragma unroll
    for (int z = 0; z < 8; ++z)
      Sk[bufI][(dg * 8 + z) * LDSTRIDE + n] = vk[z];
  };

  // Phase 1: S = q^T k  (64x64, K = N = 4096 in chunks of 32)
  const int t0 = wid * 2;  // this wave's 2 of 16 score tiles
  stage(0, 0);
  int buf = 0;
  for (int c = 0; c < 128; ++c) {
    __syncthreads();
    if (c + 1 < 128) stage(buf ^ 1, (c + 1) * 32);
#pragma unroll
    for (int s = 0; s < 2; ++s) {
      int t = t0 + s;
      int i = t >> 2, j = t & 3;
      v16bf a  = frag_a_lds(&Sq[buf][0], LDSTRIDE, i * 16, 0, lane);
      v16bf bb = frag_b_lds(&Sk[buf][0], LDSTRIDE, j * 16, 0, lane);
      acc[s] = wmma_bf16(a, bb, acc[s]);
    }
    buf ^= 1;
  }

  // Phase 2: scale + softmax over e
  __syncthreads();
  const int mh = (lane & 16) ? 8 : 0;
  const int nc = lane & 15;
  const float SCALE = 0.125f;  // 64^-0.5
#pragma unroll
  for (int s = 0; s < 2; ++s) {
    int t = t0 + s;
    int i = t >> 2, j = t & 3;
#pragma unroll
    for (int e = 0; e < 8; ++e)
      Sf[i * 16 + e + mh][j * 16 + nc] = acc[s][e] * SCALE;
  }
  __syncthreads();
  if (tid < 64) {
    float m = -1e30f;
    for (int e = 0; e < 64; ++e) m = fmaxf(m, Sf[tid][e]);
    float ssum = 0.f;
    for (int e = 0; e < 64; ++e) {
      float ex = __expf(Sf[tid][e] - m);
      Sf[tid][e] = ex;
      ssum += ex;
    }
    float inv = 1.f / ssum;
    for (int e = 0; e < 64; ++e) Sab[tid * 72 + e] = f2bf(Sf[tid][e] * inv);
  }
  __syncthreads();

  // Phase 3: out[n,d] = sum_e attn[d,e] * v[n,e]  ->  out = V @ attn^T
  v16bf bfr[2][4];
#pragma unroll
  for (int kc = 0; kc < 2; ++kc)
#pragma unroll
    for (int j = 0; j < 4; ++j)
      bfr[kc][j] = frag_b_lds(Sab, 72, j * 16, kc * 32, lane);

  for (int mt = wid; mt < 256; mt += 8) {
    int m0 = mt * 16;
    v8f o[4];
#pragma unroll
    for (int j = 0; j < 4; ++j) o[j] = zf;
#pragma unroll
    for (int kc = 0; kc < 2; ++kc) {
      const unsigned short* p =
          vbp + (size_t)(m0 + (lane & 15)) * 1536 + kc * 32 + ((lane & 16) ? 8 : 0);
      uint4 lo = *(const uint4*)p;
      uint4 hi = *(const uint4*)(p + 16);
      v16bf a = make_frag(lo, hi);
#pragma unroll
      for (int j = 0; j < 4; ++j)
        o[j] = wmma_bf16(a, bfr[kc][j], o[j]);
    }
#pragma unroll
    for (int j = 0; j < 4; ++j) {
      int d = j * 16 + nc;
#pragma unroll
      for (int e = 0; e < 8; ++e) {
        int n = m0 + e + mh;
        qb[(size_t)n * 1536 + d] = f2bf(o[j][e]);
      }
    }
  }
}

// ---------------------------------------------------------------------------
// Launch
// ---------------------------------------------------------------------------
extern "C" void kernel_launch(void* const* d_in, const int* in_sizes, int n_in,
                              void* d_out, int out_size, void* d_ws, size_t ws_size,
                              hipStream_t stream) {
  (void)in_sizes; (void)n_in; (void)out_size; (void)ws_size;
  const float* x   = (const float*)d_in[0];
  const float* Wq  = (const float*)d_in[1];
  const float* Wkv = (const float*)d_in[2];
  const float* Wp  = (const float*)d_in[3];
  const float* bp  = (const float*)d_in[4];
  float* out = (float*)d_out;

  char* ws = (char*)d_ws;
  const size_t XB  = (size_t)32768 * 512;    // x in bf16 (elements)
  const size_t QKV = (size_t)32768 * 1536;   // q|k|v in bf16
  unsigned short* xb   = (unsigned short*)ws;
  unsigned short* qkv  = (unsigned short*)(ws + XB * 2);
  unsigned short* wqkv = (unsigned short*)(ws + (XB + QKV) * 2);
  unsigned short* wp   = wqkv + 1536 * 512;
  // total workspace = 136,314,880 bytes (~130 MiB)

  // 1) quantize inputs/weights to bf16
  cvt_f32_bf16_kernel<<<(int)(XB / 2048), 256, 0, stream>>>(x, xb, (int)XB);
  cvt_f32_bf16_kernel<<<512 * 512 / 2048, 256, 0, stream>>>(Wq, wqkv, 512 * 512);
  cvt_f32_bf16_kernel<<<1024 * 512 / 2048, 256, 0, stream>>>(Wkv, wqkv + 512 * 512, 1024 * 512);
  cvt_f32_bf16_kernel<<<512 * 512 / 2048, 256, 0, stream>>>(Wp, wp, 512 * 512);

  // 2) fused q/kv projection: [32768,512] @ [1536,512]^T -> qkv bf16
  gemm_bf16_kernel<<<dim3(12, 256), 256, 0, stream>>>(
      xb, 512, wqkv, 512, qkv, nullptr, 1536, nullptr);

  // 3) channel attention per (b,h); writes output into the q columns
  chan_attn_kernel<<<64, 256, 0, stream>>>(qkv);

  // 4) output projection: [32768,512](lda=1536) @ [512,512]^T + bias -> f32 out
  gemm_bf16_kernel<<<dim3(4, 256), 256, 0, stream>>>(
      qkv, 1536, wp, 512, nullptr, out, 512, bp);
}
